// PointTransformerLayer_2173253452354
// MI455X (gfx1250) — matrix operand order repaired and checked by reading, not compiled, not run on hardware
//
#include <hip/hip_runtime.h>
#include <hip/hip_bf16.h>

typedef __attribute__((ext_vector_type(16))) _Float16 v16h;
typedef __attribute__((ext_vector_type(8)))  float    v8f;

#define C_   128
#define N_   4096
#define K_   16
#define NP   4          // n-values per block
#define P_   64         // columns per block = NP*K_
#define LSTR 144        // LDS row stride in halves (288B, 32B aligned)

__device__ __forceinline__ v8f wmma16(v16h a, v16h b, v8f c) {
  // D = A(16x32 f16) * B(32x16 f16) + C(16x16 f32)
  return __builtin_amdgcn_wmma_f32_16x16x32_f16(
      /*neg_a=*/false, a, /*neg_b=*/false, b,
      /*c_mod=*/(short)0, c, /*reuse_a=*/false, /*reuse_b=*/false);
}

// A fragment: W row-major [128][128] f16. contiguous k order used here, same as B:
// lanes 0-15 hold K base+0..15 (halves 0..15);
// lanes 16-31 hold base+16..31.
__device__ __forceinline__ v16h load_afrag(const _Float16* __restrict__ W,
                                           int row0, int ks, int lane) {
  const int m  = lane & 15;
  const int hi = lane >> 4;
  const _Float16* base = W + (size_t)(row0 + m) * C_ + ks * 32 + hi * 16;
  union { v16h v; uint4 u[2]; } r;
  r.u[0] = *(const uint4*)(base);
  r.u[1] = *(const uint4*)(base + 8);
  return r.v;
}

// B fragment from transposed LDS panel S[p][c]: B(32x16), lane = column p,
// lanes 0-15 hold K base+0..15, lanes 16-31 hold base+16..31 (contiguous c).
__device__ __forceinline__ v16h load_bfrag(const _Float16* S, int cb, int ks, int lane) {
  const int col = lane & 15;
  const int hi  = lane >> 4;
  return *(const v16h*)(S + (cb * 16 + col) * LSTR + ks * 32 + hi * 16);
}

__global__ void wcvt_kernel(const float* __restrict__ s0, const float* __restrict__ s1,
                            const float* __restrict__ s2, const float* __restrict__ s3,
                            const float* __restrict__ s4, _Float16* __restrict__ dst) {
  int idx = blockIdx.x * blockDim.x + threadIdx.x;   // 0 .. 5*16384-1
  int w = idx >> 14;
  int e = idx & 16383;
  const float* s = (w == 0) ? s0 : (w == 1) ? s1 : (w == 2) ? s2 : (w == 3) ? s3 : s4;
  dst[idx] = (_Float16)s[e];
}

__global__ __launch_bounds__(256)
void pt_layer_kernel(
    const float* __restrict__ feats,
    const float* __restrict__ gfeats,
    const float* __restrict__ points,
    const float* __restrict__ gpoints,
    const float* __restrict__ pe1_w, const float* __restrict__ pe1_b,
    const float* __restrict__ pe2_b,
    const float* __restrict__ q_b,
    const float* __restrict__ k_w,  const float* __restrict__ k_b,
    const float* __restrict__ v_b,
    const float* __restrict__ ae1_b,
    const float* __restrict__ ae2_b,
    const _Float16* __restrict__ wPE2,
    const _Float16* __restrict__ wQ,
    const _Float16* __restrict__ wV,
    const _Float16* __restrict__ wAE1,
    const _Float16* __restrict__ wAE2,
    float* __restrict__ out)
{
  __shared__ _Float16 sB0[P_ * LSTR];   // gf^T  -> later hid2
  __shared__ _Float16 sB1[P_ * LSTR];   // hid   -> later T
  __shared__ _Float16 sB2[P_ * LSTR];   // delta
  __shared__ float    sRel[3 * P_];
  __shared__ float    sK[NP * C_];

  const int tid  = threadIdx.x;
  const int lane = tid & 31;
  const int wv   = tid >> 5;          // wave id 0..7 (wave32)
  const int rb   = wv * 16;           // this wave's output-channel row block
  const int b    = blockIdx.x >> 10;  // gridDim.x = 4 * 1024
  const int n0   = (blockIdx.x & 1023) * NP;

  // ---------------- Phase 0: stage rel coords and gf^T (f16) ----------------
  if (tid < 3 * P_) {
    int j = tid / P_, p = tid % P_;
    int n = n0 + (p >> 4), kk = p & 15;
    size_t base = ((size_t)b * 3 + j) * N_ + n;
    sRel[tid] = points[base] - gpoints[base * K_ + kk];
  }
  {
    int c = tid >> 1, part = tid & 1;
    const float* g = gfeats + (((size_t)b * C_ + c) * N_ + n0) * K_ + part * 32;
#pragma unroll
    for (int i = 0; i < 8; ++i) {
      float4 f = ((const float4*)g)[i];
      int p = part * 32 + i * 4;
      sB0[(p + 0) * LSTR + c] = (_Float16)f.x;
      sB0[(p + 1) * LSTR + c] = (_Float16)f.y;
      sB0[(p + 2) * LSTR + c] = (_Float16)f.z;
      sB0[(p + 3) * LSTR + c] = (_Float16)f.w;
    }
  }
  __syncthreads();

  // -------- Phase 1: hid = relu(pe1 . rel + b1) -> sB1 ; kvec -> sK --------
  {
    int c = tid >> 1, part = tid & 1;
    float w0 = pe1_w[c * 3 + 0], w1 = pe1_w[c * 3 + 1], w2 = pe1_w[c * 3 + 2];
    float bb = pe1_b[c];
#pragma unroll
    for (int i = 0; i < 32; ++i) {
      int p = part * 32 + i;
      float h = fmaxf(w0 * sRel[p] + w1 * sRel[P_ + p] + w2 * sRel[2 * P_ + p] + bb, 0.f);
      sB1[p * LSTR + c] = (_Float16)h;
    }
  }
  {
    int c = tid & 127, hv = tid >> 7;
    const float* wr = k_w + (size_t)c * C_;
#pragma unroll
    for (int rep = 0; rep < 2; ++rep) {
      int np = hv * 2 + rep;
      const float* fc = feats + (size_t)b * C_ * N_ + n0 + np;
      float acc = k_b[c];
      for (int cc = 0; cc < C_; ++cc)
        acc += wr[cc] * fc[(size_t)cc * N_];
      sK[np * C_ + c] = acc;
    }
  }
  __syncthreads();

  // -------------- Phase 2: delta = pe2 x hid + b2 -> sB2 (f16) --------------
  {
    v16h a[4];
#pragma unroll
    for (int ks = 0; ks < 4; ++ks) a[ks] = load_afrag(wPE2, rb, ks, lane);
    const int col = lane & 15, hi = lane >> 4;
#pragma unroll
    for (int cb = 0; cb < NP; ++cb) {
      v8f acc = {};
#pragma unroll
      for (int ks = 0; ks < 4; ++ks)
        acc = wmma16(a[ks], load_bfrag(sB1, cb, ks, lane), acc);
      int p = cb * 16 + col;
#pragma unroll
      for (int j = 0; j < 8; ++j) {
        int c = rb + j + 8 * hi;
        sB2[p * LSTR + c] = (_Float16)(acc[j] + pe2_b[c]);
      }
    }
  }
  __syncthreads();

  // ---- Phase 3: q & v GEMMs (shared B frag over gf); T -> sB1; v in regs ----
  float vreg[NP][8];
  {
    v16h aq[4], av[4];
#pragma unroll
    for (int ks = 0; ks < 4; ++ks) {
      aq[ks] = load_afrag(wQ, rb, ks, lane);
      av[ks] = load_afrag(wV, rb, ks, lane);
    }
    const int col = lane & 15, hi = lane >> 4;
#pragma unroll
    for (int cb = 0; cb < NP; ++cb) {
      v8f qa = {}, va = {};
#pragma unroll
      for (int ks = 0; ks < 4; ++ks) {
        v16h bf = load_bfrag(sB0, cb, ks, lane);
        qa = wmma16(aq[ks], bf, qa);
        va = wmma16(av[ks], bf, va);
      }
      int p = cb * 16 + col;
#pragma unroll
      for (int j = 0; j < 8; ++j) {
        int c = rb + j + 8 * hi;
        float d = (float)sB2[p * LSTR + c];
        vreg[cb][j] = va[j] + v_b[c] + d;                       // v = Wv*gf + b + delta
        float t = sK[cb * C_ + c] - (qa[j] + q_b[c]) + d;       // k - q + delta
        sB1[p * LSTR + c] = (_Float16)t;
      }
    }
  }
  __syncthreads();

  // ------------ Phase 4: hid2 = relu(ae1 x T + ae1_b) -> sB0 (f16) ----------
  {
    v16h a[4];
#pragma unroll
    for (int ks = 0; ks < 4; ++ks) a[ks] = load_afrag(wAE1, rb, ks, lane);
    const int col = lane & 15, hi = lane >> 4;
#pragma unroll
    for (int cb = 0; cb < NP; ++cb) {
      v8f acc = {};
#pragma unroll
      for (int ks = 0; ks < 4; ++ks)
        acc = wmma16(a[ks], load_bfrag(sB1, cb, ks, lane), acc);
      int p = cb * 16 + col;
#pragma unroll
      for (int j = 0; j < 8; ++j) {
        int c = rb + j + 8 * hi;
        sB0[p * LSTR + c] = (_Float16)fmaxf(acc[j] + ae1_b[c], 0.f);
      }
    }
  }
  __syncthreads();

  // ---------- Phase 5: scores = ae2 x hid2 + b; softmax over k; output -------
  {
    v16h a[4];
#pragma unroll
    for (int ks = 0; ks < 4; ++ks) a[ks] = load_afrag(wAE2, rb, ks, lane);
    const int col = lane & 15, hi = lane >> 4;
#pragma unroll
    for (int cb = 0; cb < NP; ++cb) {
      v8f sa = {};
#pragma unroll
      for (int ks = 0; ks < 4; ++ks)
        sa = wmma16(a[ks], load_bfrag(sB0, cb, ks, lane), sa);
      // one col-block == one n' : its 16 columns are the K softmax elements,
      // living across lanes 0-15 (channels rb+j) and 16-31 (channels rb+8+j).
#pragma unroll
      for (int j = 0; j < 8; ++j) {
        int c = rb + j + 8 * hi;
        float s = sa[j] + ae2_b[c];
        float m = s;
#pragma unroll
        for (int msk = 1; msk < 16; msk <<= 1) m = fmaxf(m, __shfl_xor(m, msk, 16));
        float e = __expf(s - m);
        float den = e;
#pragma unroll
        for (int msk = 1; msk < 16; msk <<= 1) den += __shfl_xor(den, msk, 16);
        float num = e * vreg[cb][j];
#pragma unroll
        for (int msk = 1; msk < 16; msk <<= 1) num += __shfl_xor(num, msk, 16);
        if (col == 0)
          out[((size_t)b * C_ + c) * N_ + n0 + cb] = num / den;
      }
    }
  }
}

extern "C" void kernel_launch(void* const* d_in, const int* in_sizes, int n_in,
                              void* d_out, int out_size, void* d_ws, size_t ws_size,
                              hipStream_t stream) {
  const float* feats   = (const float*)d_in[0];
  const float* gfeats  = (const float*)d_in[1];
  const float* points  = (const float*)d_in[2];
  const float* gpoints = (const float*)d_in[3];
  const float* pe1_w = (const float*)d_in[4];
  const float* pe1_b = (const float*)d_in[5];
  const float* pe2_w = (const float*)d_in[6];
  const float* pe2_b = (const float*)d_in[7];
  const float* q_w   = (const float*)d_in[8];
  const float* q_b   = (const float*)d_in[9];
  const float* k_w   = (const float*)d_in[10];
  const float* k_b   = (const float*)d_in[11];
  const float* v_w   = (const float*)d_in[12];
  const float* v_b   = (const float*)d_in[13];
  const float* ae1_w = (const float*)d_in[14];
  const float* ae1_b = (const float*)d_in[15];
  const float* ae2_w = (const float*)d_in[16];
  const float* ae2_b = (const float*)d_in[17];

  _Float16* wbuf = (_Float16*)d_ws;   // [pe2 | q | v | ae1 | ae2], 16384 halves each

  wcvt_kernel<<<(5 * 16384) / 256, 256, 0, stream>>>(pe2_w, q_w, v_w, ae1_w, ae2_w, wbuf);

  pt_layer_kernel<<<4 * (N_ / NP), 256, 0, stream>>>(
      feats, gfeats, points, gpoints,
      pe1_w, pe1_b, pe2_b, q_b, k_w, k_b, v_b, ae1_b, ae2_b,
      wbuf, wbuf + 16384, wbuf + 2 * 16384, wbuf + 3 * 16384, wbuf + 4 * 16384,
      (float*)d_out);
}
